// InvolutionConv2D_40149354283369
// MI455X (gfx1250) — matrix operand order repaired and checked by reading, not compiled, not run on hardware
//
#include <hip/hip_runtime.h>
#include <hip/hip_bf16.h>

typedef __attribute__((ext_vector_type(16))) _Float16 v16h;
typedef __attribute__((ext_vector_type(8)))  float    v8f;

#define B_   8
#define C_   128
#define H_   112
#define W_   112
#define G_   4
#define CG   32            // C / G
#define T_   9             // K*K taps
#define OCG  32            // OC / G
#define WAVES 7
#define TPB  (WAVES * 32)
#define ROWS 4             // output rows per workgroup
#define SLAB_R (ROWS + 2)  // 6 rows incl. 1-row halo each side
#define SLAB_C (W_ + 2)    // 114 (1-col halo each side)

// 16-bit A-matrix 16x32 layout (contiguous, matching B): half e of lane l holds K =
//   e + 16*(l>=16), M = l & 15.
__device__ __forceinline__ int a_kof(int e, int lane) {
    return e + ((lane >= 16) ? 16 : 0);
}

__global__ __launch_bounds__(TPB)
void involution_fused_kernel(const float* __restrict__ x,
                             const float* __restrict__ w_kg,
                             const float* __restrict__ b_kg,
                             const float* __restrict__ w_pw,
                             const float* __restrict__ b_pw,
                             float* __restrict__ out) {
    // x slab for this (b, g, 4-row block): 6 rows x 114 cols x 32 ch, f16,
    // channel-innermost so one B-fragment = 2 contiguous ds_load_b128 / lane.
    __shared__ __align__(32) _Float16 sl[SLAB_R * SLAB_C * CG];   // 43776 B
    // per-wave staging of the dynamically generated 9x16 kernel (padded 16x16)
    __shared__ float kt[WAVES][16][16];                            // 7168 B

    const int tid  = threadIdx.x;
    const int lane = tid & 31;
    const int wv   = tid >> 5;

    const int bb = blockIdx.x / (G_ * (H_ / ROWS));
    const int g  = (blockIdx.x / (H_ / ROWS)) % G_;
    const int h0 = (blockIdx.x % (H_ / ROWS)) * ROWS;

    // ---- Stage 1: cooperative slab load (global-coalesced, zero halo) ----
    for (int idx = tid; idx < CG * SLAB_R * SLAB_C; idx += TPB) {
        const int c   = idx / (SLAB_R * SLAB_C);
        const int rr  = (idx / SLAB_C) % SLAB_R;
        const int col = idx % SLAB_C;
        const int xr = h0 + rr - 1;
        const int xc = col - 1;
        float v = 0.0f;
        if (xr >= 0 && xr < H_ && xc >= 0 && xc < W_)
            v = x[((size_t)(bb * C_ + g * CG + c) * H_ + xr) * W_ + xc];
        sl[(rr * SLAB_C + col) * CG + c] = (_Float16)v;
    }

    // ---- One-time per-wave constant fragments (A matrices, biases) ----
    const int m = lane & 15;                  // A-fragment row (output chan)
    const int mhi = (lane >> 4) << 3;         // C/D layout: M = r + 8*(l>=16)

    // A fragments for kernel-gen GEMM: w_kg[g*9+m, k, tap], one per tap.
    v16h akg[T_];
#pragma unroll
    for (int t = 0; t < T_; ++t) {
        v16h a;
#pragma unroll
        for (int e = 0; e < 16; ++e) {
            const int k = a_kof(e, lane);
            float wval = 0.0f;
            if (m < T_)
                wval = w_kg[((size_t)(g * T_ + m) * CG + k) * T_ + t];
            a[e] = (_Float16)wval;
        }
        akg[t] = a;
    }
    // A fragments for pointwise GEMM: two 16-row OC tiles of w_pw[oc, k].
    v16h apw[2];
#pragma unroll
    for (int oct = 0; oct < 2; ++oct) {
        v16h a;
#pragma unroll
        for (int e = 0; e < 16; ++e) {
            const int k = a_kof(e, lane);
            a[e] = (_Float16)w_pw[(size_t)(g * OCG + oct * 16 + m) * CG + k];
        }
        apw[oct] = a;
    }
    // Per-lane bias values matching the f32 C/D layout rows.
    float bkgv[8], bpwv[2][8];
#pragma unroll
    for (int r = 0; r < 8; ++r) {
        const int M = r + mhi;
        bkgv[r] = (M < T_) ? b_kg[g * T_ + M] : 0.0f;
        bpwv[0][r] = b_pw[g * OCG + M];
        bpwv[1][r] = b_pw[g * OCG + 16 + M];
    }

    __syncthreads();   // slab visible to all waves

    // ---- Per-wave 16-pixel column block, looped over ROWS output rows ----
    const int w0  = wv * 16;
    const int pix = lane & 15;                 // N index
    const int chb = (lane >> 4) << 4;          // B layout: K = e + 16*(l>=16)

    for (int r0 = 0; r0 < ROWS; ++r0) {
        // B fragments, one per tap:
        // B[k][n] = x[ch=k, row h0+r0+di-1, col w0+n+dj-1].
        v16h bx[T_];
#pragma unroll
        for (int di = 0; di < 3; ++di) {
#pragma unroll
            for (int dj = 0; dj < 3; ++dj) {
                const int off =
                    (((r0 + di) * SLAB_C + (w0 + pix + dj)) * CG + chb);
                bx[di * 3 + dj] = *(const v16h*)&sl[off]; // 2x ds_load_b128
            }
        }

        // ---- GEMM1: dynamic kernel gen, K accumulated over 9 taps ----
        v8f kacc = {};
#pragma unroll
        for (int t = 0; t < T_; ++t) {
            kacc = __builtin_amdgcn_wmma_f32_16x16x32_f16(
                false, akg[t], false, bx[t], (short)0, kacc, false, false);
        }
        // Stage kernel (+bias) to per-wave LDS tile in M,N order.
#pragma unroll
        for (int r = 0; r < 8; ++r)
            kt[wv][r + mhi][pix] = kacc[r] + bkgv[r];
        asm volatile("s_wait_dscnt 0" ::: "memory");  // per-wave LDS RAW fence

        // ---- Stage 2: 9-tap apply, pure f16 on the in-register fragments ---
        _Float16 kvh[T_];
#pragma unroll
        for (int t = 0; t < T_; ++t)
            kvh[t] = (_Float16)kt[wv][t][pix];

        v16h bmid;
#pragma unroll
        for (int e = 0; e < 16; ++e) {
            _Float16 acc = (_Float16)0.0f;
#pragma unroll
            for (int t = 0; t < T_; ++t)
                acc += kvh[t] * bx[t][e];      // mid[c'=e+chb, pix], f16 fma
            bmid[e] = acc;
        }

        // ---- GEMM2: grouped 1x1 pointwise conv + bias + store ----
        const size_t obase = ((size_t)bb * C_ + g * OCG) * H_ * W_
                           + (size_t)(h0 + r0) * W_ + (w0 + pix);
#pragma unroll
        for (int oct = 0; oct < 2; ++oct) {
            v8f oacc = {};
            oacc = __builtin_amdgcn_wmma_f32_16x16x32_f16(
                false, apw[oct], false, bmid, (short)0, oacc, false, false);
#pragma unroll
            for (int r = 0; r < 8; ++r) {
                const int oc = oct * 16 + r + mhi;
                out[obase + (size_t)oc * H_ * W_] = oacc[r] + bpwv[oct][r];
            }
        }
    }
}

extern "C" void kernel_launch(void* const* d_in, const int* in_sizes, int n_in,
                              void* d_out, int out_size, void* d_ws, size_t ws_size,
                              hipStream_t stream) {
    const float* x    = (const float*)d_in[0];
    const float* w_kg = (const float*)d_in[1];
    const float* b_kg = (const float*)d_in[2];
    const float* w_pw = (const float*)d_in[3];
    const float* b_pw = (const float*)d_in[4];
    float* out = (float*)d_out;
    dim3 grid(B_ * G_ * (H_ / ROWS));
    involution_fused_kernel<<<grid, TPB, 0, stream>>>(x, w_kg, b_kg, w_pw, b_pw, out);
}
